// CompressedLinear_31550829757013
// MI455X (gfx1250) — matrix operand order repaired and checked by reading, not compiled, not run on hardware
//
#include <hip/hip_runtime.h>

typedef __attribute__((ext_vector_type(16))) __bf16 v16bf;
typedef __attribute__((ext_vector_type(8)))  __bf16 v8bf;
typedef __attribute__((ext_vector_type(4)))  __bf16 v4bf;
typedef __attribute__((ext_vector_type(8)))  float  v8f;

#define LDS_STRIDE 40  // 32 bf16 payload + 8 pad => 80B row stride (16B aligned, bank-conflict free)

// ---------------- CDNA5 async copy helpers ----------------

// GLOBAL_LOAD_ASYNC_TO_LDS_B128: memory -> LDS, tracked by ASYNCcnt, no VGPR data.
// lds_addr is the LDS byte address in a VGPR (flat aperture: low 32 bits of a
// generic pointer to __shared__ memory).
__device__ __forceinline__ void async_cp16(unsigned lds_addr, const void* gptr) {
    asm volatile("global_load_async_to_lds_b128 %0, %1, off"
                 :: "v"(lds_addr), "v"((unsigned long long)(uintptr_t)gptr)
                 : "memory");
}

template<int N>
__device__ __forceinline__ void wait_asynccnt() {
    asm volatile("s_wait_asynccnt %0" :: "i"(N) : "memory");
}

// ---------------- elementwise prep kernels ----------------

__global__ void f32_to_bf16_x4(const float* __restrict__ in, __bf16* __restrict__ out, int n4) {
    int i = blockIdx.x * blockDim.x + threadIdx.x;
    if (i >= n4) return;
    float4 v = ((const float4*)in)[i];
    v4bf o;
    o.x = (__bf16)v.x; o.y = (__bf16)v.y; o.z = (__bf16)v.z; o.w = (__bf16)v.w;
    ((v4bf*)out)[i] = o;
}

__global__ void dequant_to_bf16(const int* __restrict__ q,
                                const float* __restrict__ scale,
                                const float* __restrict__ zp,
                                __bf16* __restrict__ out, int n) {
    int i = blockIdx.x * blockDim.x + threadIdx.x;
    if (i >= n) return;
    out[i] = (__bf16)(((float)q[i] - zp[0]) * scale[0]);
}

// U_s[o, r] = dequant(U[o, r]) * dequant(S[r])  — folds diagonal S into U
__global__ void dequant_fold_S(const int* __restrict__ Uq,
                               const float* __restrict__ Usc, const float* __restrict__ Uzp,
                               const int* __restrict__ Sq,
                               const float* __restrict__ Ssc, const float* __restrict__ Szp,
                               __bf16* __restrict__ out, int rank, int n) {
    int i = blockIdx.x * blockDim.x + threadIdx.x;
    if (i >= n) return;
    int r = i & (rank - 1);
    float u = ((float)Uq[i] - Uzp[0]) * Usc[0];
    float s = ((float)Sq[r] - Szp[0]) * Ssc[0];
    out[i] = (__bf16)(u * s);
}

// ---------------- tiled bf16 WMMA GEMM, async double-buffered LDS ----------------
// C[N x M] = A[N x K] @ W[M x K]^T  (+ bias), A/W row-major bf16, K % 32 == 0.
// Block: 256 threads (8 waves), 128x128 output tile. Wave (wm in 0..1, wn in 0..3)
// owns a 64x32 sub-tile = 4x2 WMMA 16x16 tiles. f32 accumulation.
// K stepped by 32 through double-buffered LDS filled with async global->LDS copies.

template<bool OUT_BF16, bool HAS_BIAS>
__global__ __launch_bounds__(256, 2)
void wmma_gemm_bf16(const __bf16* __restrict__ A, const __bf16* __restrict__ W,
                    float* __restrict__ Cf, __bf16* __restrict__ Cb,
                    const float* __restrict__ bias, int M, int K) {
    __shared__ __bf16 ldsA[2][128 * LDS_STRIDE];
    __shared__ __bf16 ldsW[2][128 * LDS_STRIDE];

    const int tid   = threadIdx.x;
    const int lane  = tid & 31;
    const int wid   = tid >> 5;
    const int wm    = wid >> 2;      // 0..1  -> 64-row slice
    const int wn    = wid & 3;       // 0..3  -> 32-col slice
    const int khalf = lane >> 4;     // 0..1
    const int l16   = lane & 15;

    const int blockRow = blockIdx.y * 128;
    const int blockCol = blockIdx.x * 128;

    // per-thread copy assignment: 2x 16B chunks of each tile (512 chunks / 256 thr)
    const int r0 = tid >> 2,           c0 = tid & 3;          // chunk tid
    const int r1 = (tid + 256) >> 2,   c1 = (tid + 256) & 3;  // chunk tid+256
    const unsigned la0 = (r0 * LDS_STRIDE + c0 * 8) * 2;      // byte offsets in tile
    const unsigned la1 = (r1 * LDS_STRIDE + c1 * 8) * 2;
    const __bf16* gA0 = A + (size_t)(blockRow + r0) * K + c0 * 8;
    const __bf16* gA1 = A + (size_t)(blockRow + r1) * K + c1 * 8;
    const __bf16* gW0 = W + (size_t)(blockCol + r0) * K + c0 * 8;
    const __bf16* gW1 = W + (size_t)(blockCol + r1) * K + c1 * 8;
    const unsigned baseA0 = (unsigned)(uintptr_t)&ldsA[0][0];
    const unsigned baseA1 = (unsigned)(uintptr_t)&ldsA[1][0];
    const unsigned baseW0 = (unsigned)(uintptr_t)&ldsW[0][0];
    const unsigned baseW1 = (unsigned)(uintptr_t)&ldsW[1][0];

    const v8f vzero = {};
    v8f acc[4][2];
#pragma unroll
    for (int t = 0; t < 4; ++t)
#pragma unroll
        for (int u = 0; u < 2; ++u) acc[t][u] = vzero;

    const int nK = K >> 5;

    // prologue: tile 0 -> buffer 0 (4 async b128 copies per thread)
    async_cp16(baseA0 + la0, gA0);
    async_cp16(baseA0 + la1, gA1);
    async_cp16(baseW0 + la0, gW0);
    async_cp16(baseW0 + la1, gW1);

    for (int kt = 0; kt < nK; ++kt) {
        const int p = kt & 1;
        if (kt + 1 < nK) {
            const int k1 = (kt + 1) << 5;
            const unsigned bA = p ? baseA0 : baseA1;
            const unsigned bW = p ? baseW0 : baseW1;
            async_cp16(bA + la0, gA0 + k1);
            async_cp16(bA + la1, gA1 + k1);
            async_cp16(bW + la0, gW0 + k1);
            async_cp16(bW + la1, gW1 + k1);
            wait_asynccnt<4>();   // tile kt's 4 copies done (async loads complete in order)
        } else {
            wait_asynccnt<0>();
        }
        __syncthreads();          // all waves' copies for tile kt landed

        const __bf16* bufA = ldsA[p];
        const __bf16* bufW = ldsW[p];

        // A fragments: lane l holds row (l&15); lanes 0-15 carry K {0..7,16..23},
        // lanes 16-31 carry K {8..15,24..31}  (ISA 7.12.2 bf16 16x32 layout)
        v16bf aF[4];
#pragma unroll
        for (int t = 0; t < 4; ++t) {
            int row = wm * 64 + t * 16 + l16;
            v8bf lo = *(const v8bf*)&bufA[row * LDS_STRIDE + khalf * 8];
            v8bf hi = *(const v8bf*)&bufA[row * LDS_STRIDE + 16 + khalf * 8];
            aF[t] = __builtin_shufflevector(lo, hi, 0,1,2,3,4,5,6,7,8,9,10,11,12,13,14,15);
        }
        // B fragments: lane l holds col (l&15); lanes 0-15 K {0..7,16..23}, lanes 16-31 K {8..15,24..31}
        v16bf bF[2];
#pragma unroll
        for (int u = 0; u < 2; ++u) {
            int row = wn * 32 + u * 16 + l16;
            v8bf lo = *(const v8bf*)&bufW[row * LDS_STRIDE + khalf * 8];
            v8bf hi = *(const v8bf*)&bufW[row * LDS_STRIDE + 16 + khalf * 8];
            bF[u] = __builtin_shufflevector(lo, hi, 0,1,2,3,4,5,6,7,8,9,10,11,12,13,14,15);
        }

#pragma unroll
        for (int t = 0; t < 4; ++t)
#pragma unroll
            for (int u = 0; u < 2; ++u)
                acc[t][u] = __builtin_amdgcn_wmma_f32_16x16x32_bf16(
                    false, aF[t], false, bF[u], (short)0, acc[t][u], false, false);

        __syncthreads();          // buffer p free for reuse at iteration kt+1
    }

    // writeback: C/D f32 layout — VGPR v, lanes 0-15 row v, lanes 16-31 row v+8
#pragma unroll
    for (int t = 0; t < 4; ++t) {
#pragma unroll
        for (int u = 0; u < 2; ++u) {
            int row0 = blockRow + wm * 64 + t * 16 + 8 * khalf;
            int col  = blockCol + wn * 32 + u * 16 + l16;
            float b = HAS_BIAS ? bias[col] : 0.0f;
#pragma unroll
            for (int v = 0; v < 8; ++v) {
                float val = acc[t][u][v] + b;
                size_t idx = (size_t)(row0 + v) * M + col;
                if (OUT_BF16) Cb[idx] = (__bf16)val;
                else          Cf[idx] = val;
            }
        }
    }
}

// ---------------- launcher ----------------

extern "C" void kernel_launch(void* const* d_in, const int* in_sizes, int n_in,
                              void* d_out, int out_size, void* d_ws, size_t ws_size,
                              hipStream_t stream) {
    (void)in_sizes; (void)n_in; (void)out_size; (void)ws_size;

    const float* x    = (const float*)d_in[0];
    const int*   Uq   = (const int*)d_in[1];
    const float* Usc  = (const float*)d_in[2];
    const float* Uzp  = (const float*)d_in[3];
    const int*   Sq   = (const int*)d_in[4];
    const float* Ssc  = (const float*)d_in[5];
    const float* Szp  = (const float*)d_in[6];
    const int*   Vq   = (const int*)d_in[7];
    const float* Vsc  = (const float*)d_in[8];
    const float* Vzp  = (const float*)d_in[9];
    const float* bias = (const float*)d_in[10];
    float* y = (float*)d_out;

    const int N = 4 * 2048, IN = 4096, R = 1024, OUT = 4096;

    char* ws = (char*)d_ws;
    __bf16* xb = (__bf16*)ws;                                   // N*IN bf16
    __bf16* vh = (__bf16*)(ws + (size_t)N * IN * 2);            // R*IN bf16
    __bf16* us = (__bf16*)(ws + ((size_t)N * IN + (size_t)R * IN) * 2);
    __bf16* h  = (__bf16*)(ws + ((size_t)N * IN + (size_t)R * IN + (size_t)OUT * R) * 2);

    // prep: x -> bf16, dequant Vh, dequant U with S folded in
    int n4 = (N * IN) / 4;
    f32_to_bf16_x4<<<n4 / 256, 256, 0, stream>>>(x, xb, n4);
    dequant_to_bf16<<<(R * IN) / 256, 256, 0, stream>>>(Vq, Vsc, Vzp, vh, R * IN);
    dequant_fold_S<<<(OUT * R) / 256, 256, 0, stream>>>(Uq, Usc, Uzp, Sq, Ssc, Szp, us, R, OUT * R);

    // GEMM1: h[N x R] = xb @ vh^T   (K = IN), bf16 output
    dim3 g1(R / 128, N / 128);
    wmma_gemm_bf16<true, false><<<g1, 256, 0, stream>>>(xb, vh, nullptr, h, nullptr, R, IN);

    // GEMM2: y[N x OUT] = h @ us^T + bias  (K = R), f32 output
    dim3 g2(OUT / 128, N / 128);
    wmma_gemm_bf16<false, true><<<g2, 256, 0, stream>>>(h, us, y, nullptr, bias, OUT, R);
}
